// LatticeLSTMCell_31628139168217
// MI455X (gfx1250) — compile-verified, matrix operand order checked by reading, not run on hardware
//
#include <hip/hip_runtime.h>

// ---------------------------------------------------------------------------
// LatticeLSTMCell for MI455X (gfx1250, wave32, WMMA).
//
// pre = X[65536,256] @ W[256,1024] + bias ; f-gate columns (256..511) are dead
// (c0 == 0), so we compute only i/g/o -> effective GEMM 65536 x 768 x 256.
// bf16 WMMA (v_wmma_f32_16x16x32_bf16) with f32 accumulate; fused
// sigmoid/tanh epilogue writes edge_h/edge_c once; tiny gather-mean kernel
// produces node_h/node_c (reads hit the 192MB L2).
//
// Round-1 disasm showed the fp32->bf16 RNE conversion dominating VALU
// (bfe+add3+mov_b16 per element).  Now: round-half-up + v_perm_b32 packs two
// floats -> one bf16x2 dword in 3 VALU ops, and the fragment is built as a
// uint32x8 vector (no 16-bit register shuffles).  Epilogue uses one exp for
// sigmoid(o)*tanh(o) (shared e = exp(-o)), three exps per h total.
// ---------------------------------------------------------------------------

typedef __attribute__((ext_vector_type(16))) __bf16          v16bf;
typedef __attribute__((ext_vector_type(8)))  unsigned int    v8u;
typedef __attribute__((ext_vector_type(8)))  float           v8f;

#define D_DIM   256
#define H_DIM   256
#define FOURH   1024
#define B_SZ    8
#define T_SZ    2048
#define K_EDGES 4
#define E_SZ    (T_SZ * K_EDGES)      // 8192
#define M_TOTAL (B_SZ * E_SZ)         // 65536
#define KCHUNKS (D_DIM / 32)          // 8

// Exact fp32 -> bf16 RNE (cold path: weight prep only)
__device__ __forceinline__ unsigned short f2bf_rne(float f) {
    unsigned u = __builtin_bit_cast(unsigned, f);
    unsigned r = u + 0x7FFFu + ((u >> 16) & 1u);
    return (unsigned short)(r >> 16);
}

// Hot path: two fp32 -> packed bf16x2, round-half-up + v_perm_b32.
// v_perm data = {S0(hi dword), S1(lo dword)}; sel 0x07060302 picks
// {hi[3],hi[2],lo[3],lo[2]} -> {bf16(hi), bf16(lo)}.
__device__ __forceinline__ unsigned pack_bf16x2(float lo, float hi) {
    unsigned a = __builtin_bit_cast(unsigned, lo) + 0x8000u;
    unsigned b = __builtin_bit_cast(unsigned, hi) + 0x8000u;
    return __builtin_amdgcn_perm(b, a, 0x07060302u);
}

// ---------------------------------------------------------------------------
// Repack W[256,1024] (fp32, row-major over K) into bf16 B-fragments.
// Fragment (nTile, kc) = B tile 32(K) x 16(N); lane L holds column
// n = nTile*16 + (L&15), K = kc*32 + (L>>4)*16 .. +15 as 16 contiguous bf16.
// 64 nTiles x 8 kc = 512 fragments x 1KB = 512KB workspace.
// ---------------------------------------------------------------------------
__global__ void __launch_bounds__(256)
prep_w(const float* __restrict__ W, unsigned short* __restrict__ wfrag) {
    int gid  = blockIdx.x * blockDim.x + threadIdx.x;   // 0 .. 16383
    int lane = gid & 31;
    int frag = gid >> 5;                                // 0 .. 511
    int kc    = frag & (KCHUNKS - 1);
    int nTile = frag >> 3;
    int n  = nTile * 16 + (lane & 15);
    int k0 = kc * 32 + (lane >> 4) * 16;
    unsigned short* dst = wfrag + (size_t)frag * 512 + (size_t)lane * 16;
#pragma unroll
    for (int j = 0; j < 16; ++j)
        dst[j] = f2bf_rne(W[(size_t)(k0 + j) * FOURH + n]);
}

// A-fragment 16x32 bf16: lane L holds row (L&15), K = k0 + (L>>4)*16 .. +15.
// 16 fp32 loads (compiler combines to 4x global_load_b128) -> 8 packed dwords.
__device__ __forceinline__ v16bf load_a_frag(const float* __restrict__ p) {
    v8u t;
#pragma unroll
    for (int j = 0; j < 8; ++j)
        t[j] = pack_bf16x2(p[2 * j], p[2 * j + 1]);
    return __builtin_bit_cast(v16bf, t);
}

__device__ __forceinline__ v16bf load_b_frag(const unsigned short* __restrict__ p) {
    return __builtin_bit_cast(v16bf, *(const v8u*)p);
}

__device__ __forceinline__ float clamp30(float x) {
    return fminf(fmaxf(x, -30.0f), 30.0f);
}

// ---------------------------------------------------------------------------
// GEMM + fused LSTM-gate epilogue.
// Block = 256 threads = 8 waves; wave owns rows [mBase, mBase+32) and 16
// h-columns.  6 f32 accumulators: (i,g,o) x 2 row-tiles; 48 v_wmma total.
// grid = (M_TOTAL/256, H_DIM/16) = (256, 16).
// ---------------------------------------------------------------------------
__global__ void __launch_bounds__(256)
gates_gemm(const float* __restrict__ A,            // [65536, 256]
           const unsigned short* __restrict__ wfrag,
           const float* __restrict__ bias,         // [1024]
           float* __restrict__ edge_h,             // [65536, 256]
           float* __restrict__ edge_c) {
    const int lane     = threadIdx.x & 31;
    const int wave     = threadIdx.x >> 5;
    const int hTile    = blockIdx.y;
    const int mBase    = blockIdx.x * 256 + wave * 32;
    const int laneHalf = lane >> 4;
    const int laneMod  = lane & 15;

    v8f ai0 = {}, ag0 = {}, ao0 = {};
    v8f ai1 = {}, ag1 = {}, ao1 = {};

    // W-column tiles: i -> nTile hTile, g -> 32+hTile, o -> 48+hTile
    const unsigned short* wfi = wfrag + ((size_t)(hTile)      * KCHUNKS) * 512 + (size_t)lane * 16;
    const unsigned short* wfg = wfrag + ((size_t)(32 + hTile) * KCHUNKS) * 512 + (size_t)lane * 16;
    const unsigned short* wfo = wfrag + ((size_t)(48 + hTile) * KCHUNKS) * 512 + (size_t)lane * 16;

    const float* arow0 = A + (size_t)(mBase + laneMod) * D_DIM + laneHalf * 16;
    const float* arow1 = arow0 + (size_t)16 * D_DIM;

#pragma unroll
    for (int kc = 0; kc < KCHUNKS; ++kc) {
        const int koff = kc * 32;
        if (kc + 1 < KCHUNKS) {                       // global_prefetch_b8
            __builtin_prefetch(arow0 + koff + 32, 0, 1);
            __builtin_prefetch(arow1 + koff + 32, 0, 1);
        }
        v16bf a0 = load_a_frag(arow0 + koff);
        v16bf a1 = load_a_frag(arow1 + koff);
        v16bf bi = load_b_frag(wfi + (size_t)kc * 512);
        v16bf bg = load_b_frag(wfg + (size_t)kc * 512);
        v16bf bo = load_b_frag(wfo + (size_t)kc * 512);

        ai0 = __builtin_amdgcn_wmma_f32_16x16x32_bf16(false, a0, false, bi, (short)0, ai0, false, false);
        ag0 = __builtin_amdgcn_wmma_f32_16x16x32_bf16(false, a0, false, bg, (short)0, ag0, false, false);
        ao0 = __builtin_amdgcn_wmma_f32_16x16x32_bf16(false, a0, false, bo, (short)0, ao0, false, false);
        ai1 = __builtin_amdgcn_wmma_f32_16x16x32_bf16(false, a1, false, bi, (short)0, ai1, false, false);
        ag1 = __builtin_amdgcn_wmma_f32_16x16x32_bf16(false, a1, false, bg, (short)0, ag1, false, false);
        ao1 = __builtin_amdgcn_wmma_f32_16x16x32_bf16(false, a1, false, bo, (short)0, ao1, false, false);
    }

    // Epilogue.  C/D layout: vgpr r, lanes 0-15 -> m = r, lanes 16-31 ->
    // m = 8+r; n = laneMod.
    const int   hcol = hTile * 16 + laneMod;
    const float bi_  = bias[hcol];
    const float bg_  = bias[512 + hcol];
    const float bo_  = bias[768 + hcol];

#pragma unroll
    for (int r = 0; r < 8; ++r) {
        const int m0 = mBase + laneHalf * 8 + r;       // row-tile 0
        const int m1 = m0 + 16;                        // row-tile 1
#pragma unroll
        for (int tile = 0; tile < 2; ++tile) {
            const float iv = clamp30((tile ? ai1[r] : ai0[r]) + bi_);
            const float gv = clamp30((tile ? ag1[r] : ag0[r]) + bg_);
            const float ov = clamp30((tile ? ao1[r] : ao0[r]) + bo_);
            const int   m  = tile ? m1 : m0;

            // edge_c = sigmoid(i) * tanh(g)           (2 exps)
            const float ei = __expf(-iv);
            const float si = 1.0f / (1.0f + ei);
            const float eg = __expf(-2.0f * gv);
            const float tg = (1.0f - eg) / (1.0f + eg);
            edge_c[(size_t)m * H_DIM + hcol] = si * tg;

            // edge_h = sigmoid(o) * tanh(o), shared e = exp(-o)   (1 exp)
            const float eo  = __expf(-ov);
            const float so  = 1.0f / (1.0f + eo);
            const float eo2 = eo * eo;
            const float to  = (1.0f - eo2) / (1.0f + eo2);
            edge_h[(size_t)m * H_DIM + hcol] = so * to;
        }
    }
}

// ---------------------------------------------------------------------------
// node_{h,c}[b,t,:] = mean over K incoming edges of edge_{h,c}[b, prev, :]
// One block per (t, b); 256 threads = one h element each.
// ---------------------------------------------------------------------------
__global__ void __launch_bounds__(256)
gather_mean(const float* __restrict__ edge_h, const float* __restrict__ edge_c,
            const int* __restrict__ prev_idx,
            float* __restrict__ node_h, float* __restrict__ node_c) {
    const int t = blockIdx.x;
    const int b = blockIdx.y;
    const int h = threadIdx.x;
    const int i0 = prev_idx[t * K_EDGES + 0];
    const int i1 = prev_idx[t * K_EDGES + 1];
    const int i2 = prev_idx[t * K_EDGES + 2];
    const int i3 = prev_idx[t * K_EDGES + 3];
    const float* eh = edge_h + (size_t)b * E_SZ * H_DIM;
    const float* ec = edge_c + (size_t)b * E_SZ * H_DIM;
    float sh = eh[(size_t)i0 * H_DIM + h] + eh[(size_t)i1 * H_DIM + h]
             + eh[(size_t)i2 * H_DIM + h] + eh[(size_t)i3 * H_DIM + h];
    float sc = ec[(size_t)i0 * H_DIM + h] + ec[(size_t)i1 * H_DIM + h]
             + ec[(size_t)i2 * H_DIM + h] + ec[(size_t)i3 * H_DIM + h];
    const size_t o = ((size_t)b * T_SZ + t) * H_DIM + h;
    node_h[o] = sh * 0.25f;
    node_c[o] = sc * 0.25f;
}

extern "C" void kernel_launch(void* const* d_in, const int* in_sizes, int n_in,
                              void* d_out, int out_size, void* d_ws, size_t ws_size,
                              hipStream_t stream) {
    const float* edge_input = (const float*)d_in[0];   // [8, 8192, 256]
    const float* weight_ih  = (const float*)d_in[1];   // [256, 1024]
    // d_in[2] = weight_hh: provably dead in the reference numerics
    const float* bias       = (const float*)d_in[3];   // [1024]
    const int*   prev_idx   = (const int*)d_in[4];     // [2048, 4]

    float* out    = (float*)d_out;
    float* node_h = out;
    float* node_c = node_h + (size_t)B_SZ * T_SZ * H_DIM;
    float* edge_h = node_c + (size_t)B_SZ * T_SZ * H_DIM;
    float* edge_c = edge_h + (size_t)B_SZ * E_SZ * H_DIM;

    unsigned short* wfrag = (unsigned short*)d_ws;     // 512 KB of scratch

    prep_w<<<64, 256, 0, stream>>>(weight_ih, wfrag);

    dim3 ggrid(M_TOTAL / 256, H_DIM / 16);             // (256, 16)
    gates_gemm<<<ggrid, 256, 0, stream>>>(edge_input, wfrag, bias, edge_h, edge_c);

    dim3 ngrid(T_SZ, B_SZ);                            // (2048, 8)
    gather_mean<<<ngrid, 256, 0, stream>>>(edge_h, edge_c, prev_idx, node_h, node_c);
}
